// Attention_7181185319444
// MI455X (gfx1250) — matrix operand order repaired and checked by reading, not compiled, not run on hardware
//
#include <hip/hip_runtime.h>

typedef __attribute__((ext_vector_type(16))) _Float16 v16h;
typedef __attribute__((ext_vector_type(8)))  _Float16 v8h;
typedef __attribute__((ext_vector_type(8)))  float    v8f;
typedef __attribute__((ext_vector_type(4)))  float    f4;
typedef __attribute__((ext_vector_type(4)))  int      v4i;

#define SD   64      // head dim
#define SEQ  2048
#define QT   128     // q rows per workgroup (2 x 16 per wave)
#define KT   32      // kv rows per block
#define NBH  64      // B*H

__device__ __forceinline__ v8h cvt8(f4 a, f4 b) {
  v8h r;
  r[0] = (_Float16)a[0]; r[1] = (_Float16)a[1];
  r[2] = (_Float16)a[2]; r[3] = (_Float16)a[3];
  r[4] = (_Float16)b[0]; r[5] = (_Float16)b[1];
  r[6] = (_Float16)b[2]; r[7] = (_Float16)b[3];
  return r;
}

__device__ __forceinline__ v16h cat16(v8h lo, v8h hv) {
  return __builtin_shufflevector(lo, hv, 0,1,2,3,4,5,6,7,8,9,10,11,12,13,14,15);
}

// xor-16 lane swap in wave32 as a single VALU op (identity selects)
__device__ __forceinline__ int xor16_i(int x) {
  return __builtin_amdgcn_permlanex16(x, x, 0x76543210, 0xfedcba98, false, false);
}
__device__ __forceinline__ float xor16_f(float x) {
  return __builtin_bit_cast(float, xor16_i(__builtin_bit_cast(int, x)));
}

__global__ __launch_bounds__(128)
void fa_fwd_f16wmma(const float* __restrict__ q, const float* __restrict__ k,
                    const float* __restrict__ v, const int* __restrict__ amask,
                    float* __restrict__ out) {
  // Double-buffered stages: K row-major f16 (A of S^T = K Q^T),
  // V transposed (A of O^T = V^T P^T).
  __shared__ __align__(16) _Float16 Ks[2][KT][72];   // 144B rows
  __shared__ __align__(16) _Float16 Vt[2][SD][40];   // 80B rows

  const int tid  = threadIdx.x;
  const int lane = tid & 31;
  const int wid  = tid >> 5;
  const int l16  = lane & 15;
  const int hih  = lane >> 4;

  const int qt = blockIdx.x;
  const int bh = blockIdx.y;
  const int bb = bh >> 4;
  const int q0 = qt * QT;
  const int qw0 = q0 + wid * 32;       // this wave's two 16-row q tiles
  int qwu[2]; qwu[0] = qw0; qwu[1] = qw0 + 16;
  int qiu[2]; qiu[0] = qw0 + l16; qiu[1] = qw0 + 16 + l16;

  const size_t base = (size_t)bh * SEQ * SD;

  const float SL = 0.125f * 1.44269504088896340736f;  // scale * log2(e)

  // ---- B operands: Q^T per tile, PRE-SCALED by scale*log2(e) so the S^T
  //      WMMA output is already in exp2 domain (kills per-block pk_muls).
  v16h bq[2][2];
  #pragma unroll
  for (int u = 0; u < 2; ++u) {
    const float* Qr = q + base + (size_t)qiu[u] * SD;
    #pragma unroll
    for (int dc = 0; dc < 2; ++dc) {
      const f4* p = (const f4*)(Qr + dc * 32 + hih * 8);
      bq[u][dc] = cat16(cvt8(p[0] * SL, p[1] * SL), cvt8(p[4] * SL, p[5] * SL));
    }
  }

  float m[2], lsum[2];
  v8f o[2][4];
  #pragma unroll
  for (int u = 0; u < 2; ++u) {
    m[u] = -1e30f; lsum[u] = 0.0f;
    #pragma unroll
    for (int d = 0; d < 4; ++d) { v8f z = {}; o[u][d] = z; }
  }

  const int nblk = qt * 4 + 4;  // causal: kv blocks covering [0, q0+128)

  // ---- prologue: stage block 0 into buffer 0
  {
    const f4* kp = (const f4*)(k + base + (size_t)lane * SD + wid * 16);
    const f4* vp = (const f4*)(v + base + (size_t)lane * SD + wid * 16);
    f4 k0 = kp[0], k1 = kp[1], k2 = kp[2], k3 = kp[3];
    *(v8h*)&Ks[0][lane][wid * 16]     = cvt8(k0, k1);
    *(v8h*)&Ks[0][lane][wid * 16 + 8] = cvt8(k2, k3);
    f4 v0 = vp[0], v1 = vp[1], v2 = vp[2], v3 = vp[3];
    _Float16 hv[16];
    #pragma unroll
    for (int i = 0; i < 4; ++i) {
      hv[i] = (_Float16)v0[i]; hv[4 + i] = (_Float16)v1[i];
      hv[8 + i] = (_Float16)v2[i]; hv[12 + i] = (_Float16)v3[i];
    }
    #pragma unroll
    for (int i = 0; i < 16; ++i) Vt[0][wid * 16 + i][lane] = hv[i];
  }
  int amc = amask[bb * SEQ + lane];
  __syncthreads();

  for (int j = 0; j < nblk; ++j) {
    const int cur = j & 1;
    const int kv0 = j * KT;
    const bool hn = (j + 1 < nblk);

    // ---- issue next block's global loads early (latency hides under compute)
    f4 nk[4], nv[4];
    int amn = 0;
    if (hn) {
      const f4* kp = (const f4*)(k + base + (size_t)(kv0 + KT + lane) * SD + wid * 16);
      const f4* vp = (const f4*)(v + base + (size_t)(kv0 + KT + lane) * SD + wid * 16);
      #pragma unroll
      for (int i = 0; i < 4; ++i) { nk[i] = kp[i]; nv[i] = vp[i]; }
      amn = amask[bb * SEQ + kv0 + KT + lane];
      if (j + 2 < nblk) {
        __builtin_prefetch(k + base + (size_t)(kv0 + 2 * KT + lane) * SD + wid * 16, 0, 1);
        __builtin_prefetch(v + base + (size_t)(kv0 + 2 * KT + lane) * SD + wid * 16, 0, 1);
      }
    }

    const unsigned mb = (unsigned)__ballot(amc != 0);
    // ---- block fully masked for both tiles of this wave?
    if (!__builtin_amdgcn_readfirstlane((kv0 > qwu[1] + 15) ? 1 : 0)) {
      // ---- S^T = K Q^T : each Ks A-operand feeds BOTH q tiles
      v8f acc[2][2];  // [tile u][kv sub-tile t]
      #pragma unroll
      for (int u = 0; u < 2; ++u)
        #pragma unroll
        for (int t = 0; t < 2; ++t) { v8f z = {}; acc[u][t] = z; }
      #pragma unroll
      for (int t = 0; t < 2; ++t)
        #pragma unroll
        for (int dc = 0; dc < 2; ++dc) {
          const v8h* alo = (const v8h*)&Ks[cur][t * 16 + l16][dc * 32 + hih * 8];
          const v8h* ahi = (const v8h*)&Ks[cur][t * 16 + l16][dc * 32 + 16 + hih * 8];
          const v16h a = cat16(alo[0], ahi[0]);
          #pragma unroll
          for (int u = 0; u < 2; ++u)
            acc[u][t] = __builtin_amdgcn_wmma_f32_16x16x32_f16(
                false, a, false, bq[u][dc], (short)0, acc[u][t], false, false);
        }

      // ---- hoist V^T A-operands (shared by both tiles' PV)
      v16h av[4];
      #pragma unroll
      for (int dt = 0; dt < 4; ++dt) {
        const v8h* alo = (const v8h*)&Vt[cur][dt * 16 + l16][hih * 16];
        const v8h* ahi = (const v8h*)&Vt[cur][dt * 16 + l16][hih * 16 + 8];
        av[dt] = cat16(alo[0], ahi[0]);
      }

      // ---- per-tile softmax + PV (scores are already exp2-domain)
      #pragma unroll
      for (int u = 0; u < 2; ++u) {
        if (__builtin_amdgcn_readfirstlane((kv0 > qwu[u] + 15) ? 1 : 0)) continue;

        float x[2][8];
        const int fastI = ((kv0 + 31 <= qwu[u]) && (mb == 0xffffffffu)) ? 1 : 0;
        if (__builtin_amdgcn_readfirstlane(fastI)) {
          #pragma unroll
          for (int t = 0; t < 2; ++t)
            #pragma unroll
            for (int r = 0; r < 8; ++r) x[t][r] = acc[u][t][r];
        } else {
          const int thr = qiu[u] - kv0;  // kloc <= thr allowed by causality
          const unsigned cw = (thr >= 31) ? 0xffffffffu
                                          : ((thr < 0) ? 0u : ((2u << thr) - 1u));
          const unsigned aw  = mb & cw;
          const unsigned aw0 = aw >> (hih * 8);
          const unsigned aw1 = aw >> (16 + hih * 8);
          #pragma unroll
          for (int r = 0; r < 8; ++r) {
            x[0][r] = ((aw0 >> r) & 1u) ? acc[u][0][r] : -1e30f;
            x[1][r] = ((aw1 >> r) & 1u) ? acc[u][1][r] : -1e30f;
          }
        }

        const float t0 = fmaxf(fmaxf(x[0][0], x[0][1]), fmaxf(x[0][2], x[0][3]));
        const float t1 = fmaxf(fmaxf(x[0][4], x[0][5]), fmaxf(x[0][6], x[0][7]));
        const float t2 = fmaxf(fmaxf(x[1][0], x[1][1]), fmaxf(x[1][2], x[1][3]));
        const float t3 = fmaxf(fmaxf(x[1][4], x[1][5]), fmaxf(x[1][6], x[1][7]));
        float pm = fmaxf(fmaxf(t0, t1), fmaxf(t2, t3));
        pm = fmaxf(pm, xor16_f(pm));
        const float mn = fmaxf(m[u], pm);
        const float alpha = __builtin_exp2f(m[u] - mn);
        float p0[8], p1[8];
        #pragma unroll
        for (int r = 0; r < 8; ++r) {
          p0[r] = __builtin_exp2f(x[0][r] - mn);
          p1[r] = __builtin_exp2f(x[1][r] - mn);
        }
        const float s0 = (p0[0] + p0[1]) + (p0[2] + p0[3]);
        const float s1 = (p0[4] + p0[5]) + (p0[6] + p0[7]);
        const float s2 = (p1[0] + p1[1]) + (p1[2] + p1[3]);
        const float s3 = (p1[4] + p1[5]) + (p1[6] + p1[7]);
        float rs = (s0 + s1) + (s2 + s3);
        rs += xor16_f(rs);
        lsum[u] = lsum[u] * alpha + rs;
        m[u] = mn;
        #pragma unroll
        for (int d = 0; d < 4; ++d)
          #pragma unroll
          for (int r = 0; r < 8; ++r) o[u][d][r] *= alpha;

        // build B operand P^T: pack f16, swap halves with xor-16 partner
        v8h ph0, ph1;
        #pragma unroll
        for (int r = 0; r < 8; ++r) { ph0[r] = (_Float16)p0[r]; ph1[r] = (_Float16)p1[r]; }
        v4i i0 = __builtin_bit_cast(v4i, ph0);
        v4i i1 = __builtin_bit_cast(v4i, ph1);
        v4i e0, e1;
        #pragma unroll
        for (int i = 0; i < 4; ++i) { e0[i] = xor16_i(i0[i]); e1[i] = xor16_i(i1[i]); }
        const v4i blo = hih ? e1 : i0;   // halves 0-7  : kv = kv0 + hih*16 + 0..7
        const v4i bhi = hih ? i1 : e0;   // halves 8-15 : kv = kv0 + hih*16 + 8..15
        const v16h bp = cat16(__builtin_bit_cast(v8h, blo), __builtin_bit_cast(v8h, bhi));

        #pragma unroll
        for (int dt = 0; dt < 4; ++dt) {
          o[u][dt] = __builtin_amdgcn_wmma_f32_16x16x32_f16(
              false, av[dt], false, bp, (short)0, o[u][dt], false, false);
        }
      }
    }

    // ---- stage next block into the other buffer
    if (hn) {
      const int nxt = cur ^ 1;
      *(v8h*)&Ks[nxt][lane][wid * 16]     = cvt8(nk[0], nk[1]);
      *(v8h*)&Ks[nxt][lane][wid * 16 + 8] = cvt8(nk[2], nk[3]);
      _Float16 hv[16];
      #pragma unroll
      for (int i = 0; i < 4; ++i) {
        hv[i] = (_Float16)nv[0][i]; hv[4 + i] = (_Float16)nv[1][i];
        hv[8 + i] = (_Float16)nv[2][i]; hv[12 + i] = (_Float16)nv[3][i];
      }
      #pragma unroll
      for (int i = 0; i < 16; ++i) Vt[nxt][wid * 16 + i][lane] = hv[i];
      amc = amn;
    }
    __syncthreads();
  }

  // ---- epilogue: O^T lane = (d rows, q col) -> contiguous d, vector stores
  #pragma unroll
  for (int u = 0; u < 2; ++u) {
    const float inv = 1.0f / lsum[u];
    float* og = out + base + (size_t)qiu[u] * SD;
    #pragma unroll
    for (int dt = 0; dt < 4; ++dt) {
      f4 a, b;
      #pragma unroll
      for (int i = 0; i < 4; ++i) {
        a[i] = o[u][dt][i] * inv;
        b[i] = o[u][dt][4 + i] * inv;
      }
      *(f4*)(og + dt * 16 + hih * 8)     = a;
      *(f4*)(og + dt * 16 + hih * 8 + 4) = b;
    }
  }
}

extern "C" void kernel_launch(void* const* d_in, const int* in_sizes, int n_in,
                              void* d_out, int out_size, void* d_ws, size_t ws_size,
                              hipStream_t stream) {
  const float* q  = (const float*)d_in[0];
  const float* k  = (const float*)d_in[1];
  const float* v  = (const float*)d_in[2];
  const int*   am = (const int*)d_in[3];
  float* out = (float*)d_out;
  dim3 grid(SEQ / QT, NBH);
  fa_fwd_f16wmma<<<grid, dim3(128), 0, stream>>>(q, k, v, am, out);
}
